// Code2Vec_34720515620919
// MI455X (gfx1250) — matrix operand order repaired and checked by reading, not compiled, not run on hardware
//
#include <hip/hip_runtime.h>
#include <hip/hip_bf16.h>
#include <stdint.h>

// ---------------------------------------------------------------------------
// Code2Vec forward for MI455X (gfx1250, wave32, WMMA).
//   K1: convert tag_emb f32 -> bf16 (row-major, feeds WMMA B tiles directly)
//   K2: prepack W_fc into the 16x16x32 bf16 WMMA B register image
//   K3: gather + FC(384->128) via v_wmma_f32_16x16x32_bf16 + tanh + attention
//       pooling -> v_bf16 [2048x128]
//   K4: classifier GEMM pass 1: online softmax stats (max/sum) per row,
//       WMMA + wave shuffles, no logit materialization
//   K5: combine stripe partials -> row max & 1/sum
//   K6: classifier GEMM pass 2: recompute logits via WMMA, write
//       exp(l - max) / sum straight to d_out (single 412 MB write)
// ---------------------------------------------------------------------------

typedef __attribute__((ext_vector_type(16))) __bf16   v16bf;
typedef __attribute__((ext_vector_type(8)))  float    v8f;
typedef __attribute__((ext_vector_type(4)))  unsigned u32x4;

union RegB16 {           // one WMMA 16x16x32 bf16 operand (8 VGPRs)
    v16bf    v;
    unsigned u[8];
    u32x4    q[2];
};

#define Bf   2048
#define Nc   10
#define Ef   128
#define VTf  50257
#define NT   3142      // ceil(50257/16)
#define NSTR 64        // column stripes for stats pass

__device__ __forceinline__ unsigned short f2bf(float f) {
    union { float f; unsigned u; } x; x.f = f;
    unsigned r = x.u + 0x7FFFu + ((x.u >> 16) & 1u);   // round-to-nearest-even
    return (unsigned short)(r >> 16);
}

// --------------------------- K1: f32 -> bf16 pairs -------------------------
__global__ void k_cvt_bf16(const float* __restrict__ in,
                           unsigned* __restrict__ out, int n2) {
    int i = blockIdx.x * 256 + threadIdx.x;
    if (i < n2) {
        unsigned lo = f2bf(in[2 * i]);
        unsigned hi = f2bf(in[2 * i + 1]);
        out[i] = lo | (hi << 16);
    }
}

// ------------------- K2: prepack W_fc into WMMA B layout -------------------
// Load-side index: wpack[(((ks*8 + tile)*32 + lane)*8 + j]
// B 32x16 layout: lane (n,h), VGPR j holds K = 16(j/4) + 8h + 2(j%4), +1.
__global__ void k_prepack_wfc(const float* __restrict__ W,
                              unsigned* __restrict__ wp) {
    int tid  = blockIdx.x * 256 + threadIdx.x;   // 12*8*32*8 = 24576
    int j    = tid & 7;
    int lane = (tid >> 3) & 31;
    int tile = (tid >> 8) & 7;
    int ks   = tid >> 11;
    int nn = lane & 15, h = lane >> 4;
    int k0  = ks * 32 + ((j >> 2) << 4) + h * 8 + 2 * (j & 3);
    int col = tile * 16 + nn;
    unsigned lo = f2bf(W[(size_t)k0 * Ef + col]);
    unsigned hi = f2bf(W[(size_t)(k0 + 1) * Ef + col]);
    wp[tid] = lo | (hi << 16);
}

// ------ K3: gather + FC (WMMA) + tanh + attention pooling, one fn/block ----
__global__ __launch_bounds__(128)
void k_ctx_att(const int* __restrict__ v1i, const int* __restrict__ pidx,
               const int* __restrict__ v2i,
               const float* __restrict__ vemb, const float* __restrict__ pemb,
               const unsigned* __restrict__ wpack, const float* __restrict__ bfc,
               const float* __restrict__ watt, const float* __restrict__ batt,
               unsigned short* __restrict__ vout) {
    __shared__ unsigned short sctx[16 * 392];   // 16 rows x 384 bf16, padded
    __shared__ float sct[16 * 132];             // c_tilde f32, padded
    __shared__ float salog[16];
    __shared__ float satt[16];

    const int tid = threadIdx.x;
    const int b   = blockIdx.x;

    // zero the ctx tile (rows 10..15 stay zero = M-padding)
    for (int i = tid; i < 16 * 392 / 2; i += 128) ((unsigned*)sctx)[i] = 0u;
    __syncthreads();

    // gather 10 contexts: [v1 | path | v2] -> bf16 in LDS
    for (int n = 0; n < Nc; ++n) {
        int i1 = v1i[b * Nc + n], ip = pidx[b * Nc + n], i2 = v2i[b * Nc + n];
        const float* r1 = vemb + (size_t)i1 * Ef;
        const float* rp = pemb + (size_t)ip * Ef;
        const float* r2 = vemb + (size_t)i2 * Ef;
        sctx[n * 392 + tid]       = f2bf(r1[tid]);
        sctx[n * 392 + 128 + tid] = f2bf(rp[tid]);
        sctx[n * 392 + 256 + tid] = f2bf(r2[tid]);
    }
    __syncthreads();

    const int lane = tid & 31, w = tid >> 5;
    const int m = lane & 15, h = lane >> 4;
    const int t0 = 2 * w, t1 = 2 * w + 1;      // each wave: 2 column tiles

    v8f acc0 = {}; v8f acc1 = {};
    #pragma unroll
    for (int ks = 0; ks < 12; ++ks) {          // K = 384 = 12 * 32
        RegB16 a;
        #pragma unroll
        for (int j = 0; j < 4; ++j) {
            int kb = ks * 32 + 8 * h + 2 * j;
            a.u[j]     = *(const unsigned*)&sctx[m * 392 + kb];
            a.u[4 + j] = *(const unsigned*)&sctx[m * 392 + kb + 16];
        }
        RegB16 b0, b1;
        const u32x4* p0 = (const u32x4*)(wpack + (size_t)((ks * 8 + t0) * 32 + lane) * 8);
        b0.q[0] = p0[0]; b0.q[1] = p0[1];
        const u32x4* p1 = (const u32x4*)(wpack + (size_t)((ks * 8 + t1) * 32 + lane) * 8);
        b1.q[0] = p1[0]; b1.q[1] = p1[1];
        acc0 = __builtin_amdgcn_wmma_f32_16x16x32_bf16(false, a.v, false, b0.v,
                                                       (short)0, acc0, false, false);
        acc1 = __builtin_amdgcn_wmma_f32_16x16x32_bf16(false, a.v, false, b1.v,
                                                       (short)0, acc1, false, false);
    }

    // bias + tanh -> LDS (C layout: lane (n,h), VGPR r holds row 8h+r)
    {
        const int nn = lane & 15;
        #pragma unroll
        for (int r = 0; r < 8; ++r) {
            int row = 8 * h + r;
            int c0 = t0 * 16 + nn, c1 = t1 * 16 + nn;
            sct[row * 132 + c0] = tanhf(acc0[r] + bfc[c0]);
            sct[row * 132 + c1] = tanhf(acc1[r] + bfc[c1]);
        }
    }
    __syncthreads();

    // attention logits over the 10 contexts
    if (tid < Nc) {
        float al = batt[0];
        for (int e = 0; e < Ef; ++e) al += sct[tid * 132 + e] * watt[e];
        salog[tid] = al;
    }
    __syncthreads();
    if (tid == 0) {
        float mx = -1e30f;
        for (int n = 0; n < Nc; ++n) mx = fmaxf(mx, salog[n]);
        float s = 0.f;
        for (int n = 0; n < Nc; ++n) { float e = __expf(salog[n] - mx); satt[n] = e; s += e; }
        float inv = 1.f / s;
        for (int n = 0; n < Nc; ++n) satt[n] *= inv;
    }
    __syncthreads();

    // pooled code vector -> bf16
    float v = 0.f;
    for (int n = 0; n < Nc; ++n) v += satt[n] * sct[n * 132 + tid];
    vout[(size_t)b * Ef + tid] = f2bf(v);
}

// ----- K4: classifier pass 1 — online softmax stats via WMMA, no stores ----
__global__ __launch_bounds__(32)
void k_logit_stats(const unsigned short* __restrict__ vbf,
                   const unsigned short* __restrict__ tagbf,
                   float* __restrict__ pmax, float* __restrict__ psum) {
    const int stripe = blockIdx.x;         // 0..NSTR-1
    const int mtile  = blockIdx.y;         // 0..127
    const int lane = threadIdx.x;
    const int nn = lane & 15, h = lane >> 4;
    const int m = mtile * 16 + nn;

    RegB16 a[4];                            // whole K=128 of A stays in VGPRs
    #pragma unroll
    for (int ks = 0; ks < 4; ++ks) {
        const u32x4* pa = (const u32x4*)(vbf + (size_t)m * Ef + ks * 32 + 8 * h);
        a[ks].q[0] = pa[0];
        a[ks].q[1] = pa[2];                 // K = 16 + 8h + 2j half
    }

    float M[8], S[8];
    #pragma unroll
    for (int r = 0; r < 8; ++r) { M[r] = -1e30f; S[r] = 0.f; }

    for (int tile = stripe; tile < NT; tile += NSTR) {
        int t = tile * 16 + nn;
        bool valid = (t < VTf);
        RegB16 bb[4];
        if (valid) {
            #pragma unroll
            for (int ks = 0; ks < 4; ++ks) {
                const u32x4* pb = (const u32x4*)(tagbf + (size_t)t * Ef + ks * 32 + 8 * h);
                bb[ks].q[0] = pb[0]; bb[ks].q[1] = pb[2];
            }
        } else {
            u32x4 z = {0u, 0u, 0u, 0u};
            #pragma unroll
            for (int ks = 0; ks < 4; ++ks) { bb[ks].q[0] = z; bb[ks].q[1] = z; }
        }
        v8f c = {};
        #pragma unroll
        for (int ks = 0; ks < 4; ++ks)
            c = __builtin_amdgcn_wmma_f32_16x16x32_bf16(false, a[ks].v, false, bb[ks].v,
                                                        (short)0, c, false, false);
        #pragma unroll
        for (int r = 0; r < 8; ++r) {
            float cm = valid ? c[r] : -1e30f;
            cm = fmaxf(cm, __shfl_xor(cm, 1, 32));   // reduce over 16-lane half
            cm = fmaxf(cm, __shfl_xor(cm, 2, 32));
            cm = fmaxf(cm, __shfl_xor(cm, 4, 32));
            cm = fmaxf(cm, __shfl_xor(cm, 8, 32));
            float newM = fmaxf(M[r], cm);
            S[r] *= __expf(M[r] - newM);             // rescale running sum
            M[r] = newM;
            S[r] += valid ? __expf(c[r] - newM) : 0.f;
        }
    }
    #pragma unroll
    for (int r = 0; r < 8; ++r) {                    // sum over the 16 columns
        S[r] += __shfl_xor(S[r], 1, 32);
        S[r] += __shfl_xor(S[r], 2, 32);
        S[r] += __shfl_xor(S[r], 4, 32);
        S[r] += __shfl_xor(S[r], 8, 32);
    }
    if (nn == 0) {
        #pragma unroll
        for (int r = 0; r < 8; ++r) {
            int row = mtile * 16 + 8 * h + r;
            pmax[(size_t)row * NSTR + stripe] = M[r];
            psum[(size_t)row * NSTR + stripe] = S[r];
        }
    }
}

// ---------------- K5: combine stripe partials -> row max, 1/sum -----------
__global__ void k_stats_combine(const float* __restrict__ pmax,
                                const float* __restrict__ psum,
                                float* __restrict__ rmax,
                                float* __restrict__ rscale) {
    int row = blockIdx.x * 256 + threadIdx.x;
    if (row >= Bf) return;
    float M = -1e30f;
    for (int s = 0; s < NSTR; ++s) M = fmaxf(M, pmax[(size_t)row * NSTR + s]);
    float S = 0.f;
    for (int s = 0; s < NSTR; ++s)
        S += psum[(size_t)row * NSTR + s] * __expf(pmax[(size_t)row * NSTR + s] - M);
    rmax[row]   = M;
    rscale[row] = 1.f / S;
}

// -------- K6: classifier pass 2 — recompute + write normalized q -----------
__global__ __launch_bounds__(128)
void k_softmax_out(const unsigned short* __restrict__ vbf,
                   const unsigned short* __restrict__ tagbf,
                   const float* __restrict__ rmax, const float* __restrict__ rscale,
                   float* __restrict__ out) {
    const int grp   = blockIdx.x;          // 0..196 (16 tiles per block)
    const int mtile = blockIdx.y;          // 0..127
    const int w = threadIdx.x >> 5, lane = threadIdx.x & 31;
    const int nn = lane & 15, h = lane >> 4;
    const int m = mtile * 16 + nn;

    RegB16 a[4];
    #pragma unroll
    for (int ks = 0; ks < 4; ++ks) {
        const u32x4* pa = (const u32x4*)(vbf + (size_t)m * Ef + ks * 32 + 8 * h);
        a[ks].q[0] = pa[0];
        a[ks].q[1] = pa[2];
    }
    float M[8], Sc[8];
    #pragma unroll
    for (int r = 0; r < 8; ++r) {
        int row = mtile * 16 + 8 * h + r;
        M[r]  = rmax[row];
        Sc[r] = rscale[row];
    }

    #pragma unroll
    for (int i = 0; i < 4; ++i) {
        int tile = grp * 16 + w * 4 + i;   // wave-uniform
        if (tile < NT) {
            int t = tile * 16 + nn;
            bool valid = (t < VTf);
            RegB16 bb[4];
            if (valid) {
                #pragma unroll
                for (int ks = 0; ks < 4; ++ks) {
                    const u32x4* pb = (const u32x4*)(tagbf + (size_t)t * Ef + ks * 32 + 8 * h);
                    bb[ks].q[0] = pb[0]; bb[ks].q[1] = pb[2];
                }
            } else {
                u32x4 z = {0u, 0u, 0u, 0u};
                #pragma unroll
                for (int ks = 0; ks < 4; ++ks) { bb[ks].q[0] = z; bb[ks].q[1] = z; }
            }
            v8f c = {};
            #pragma unroll
            for (int ks = 0; ks < 4; ++ks)
                c = __builtin_amdgcn_wmma_f32_16x16x32_bf16(false, a[ks].v, false, bb[ks].v,
                                                            (short)0, c, false, false);
            if (valid) {
                #pragma unroll
                for (int r = 0; r < 8; ++r) {
                    size_t row = (size_t)(mtile * 16 + 8 * h + r);
                    out[row * VTf + t] = __expf(c[r] - M[r]) * Sc[r];
                }
            }
        }
    }
}

// ---------------------------------------------------------------------------
extern "C" void kernel_launch(void* const* d_in, const int* in_sizes, int n_in,
                              void* d_out, int out_size, void* d_ws, size_t ws_size,
                              hipStream_t stream) {
    const int*   v1i  = (const int*)d_in[0];
    const int*   pidx = (const int*)d_in[1];
    const int*   v2i  = (const int*)d_in[2];
    const float* vemb = (const float*)d_in[3];
    const float* pemb = (const float*)d_in[4];
    const float* temb = (const float*)d_in[5];
    const float* wfc  = (const float*)d_in[6];
    const float* bfc  = (const float*)d_in[7];
    const float* watt = (const float*)d_in[8];
    const float* batt = (const float*)d_in[9];
    float* out = (float*)d_out;

    // workspace layout (~14.6 MB total)
    char* ws = (char*)d_ws;
    unsigned short* tagbf  = (unsigned short*)(ws);              // 12,865,792 B
    unsigned*       wpack  = (unsigned*)      (ws + 12865792);   //     98,304 B
    unsigned short* vbf    = (unsigned short*)(ws + 12964096);   //    524,288 B
    float*          pmax   = (float*)         (ws + 13488384);   //    524,288 B
    float*          psum   = (float*)         (ws + 14012672);   //    524,288 B
    float*          rmax   = (float*)         (ws + 14536960);   //      8,192 B
    float*          rscale = (float*)         (ws + 14545152);   //      8,192 B

    const int n2 = VTf * Ef / 2;   // 3,216,448 bf16 pairs
    k_cvt_bf16   <<<dim3((n2 + 255) / 256), dim3(256), 0, stream>>>(temb, (unsigned*)tagbf, n2);
    k_prepack_wfc<<<dim3(96),               dim3(256), 0, stream>>>(wfc, wpack);
    k_ctx_att    <<<dim3(Bf),               dim3(128), 0, stream>>>(v1i, pidx, v2i, vemb, pemb,
                                                                    wpack, bfc, watt, batt, vbf);
    k_logit_stats<<<dim3(NSTR, 128),        dim3(32),  0, stream>>>(vbf, tagbf, pmax, psum);
    k_stats_combine<<<dim3(8),              dim3(256), 0, stream>>>(pmax, psum, rmax, rscale);
    k_softmax_out<<<dim3(197, 128),         dim3(128), 0, stream>>>(vbf, tagbf, rmax, rscale, out);
}
